// MLPBlock_69569880261192
// MI455X (gfx1250) — matrix operand order repaired and checked by reading, not compiled, not run on hardware
//
#include <hip/hip_runtime.h>
#include <hip/hip_bf16.h>
#include <math.h>

#define NEXP   32
#define KSEL   4
#define HDIM   1024
#define FDIM   1024
#define LIMITC 7.0f
#define ALPHAC 1.702f
#define EPSC   1e-5f

// Gathered A-panel row stride in LDS: 2048B row + 2 x 16B TDM pad (4 dwords per 256 dwords)
#define AROW_BYTES 2080
#define AROW_ELEM  (AROW_BYTES / 2)

typedef __attribute__((ext_vector_type(16))) __bf16 v16bf;
typedef __attribute__((ext_vector_type(8)))  __bf16 bf16x8;
typedef __attribute__((ext_vector_type(8)))  float  v8f;
typedef __attribute__((ext_vector_type(4)))  unsigned int u32x4;
typedef __attribute__((ext_vector_type(8)))  int i32x8;
typedef __attribute__((ext_vector_type(4)))  int i32x4;

__device__ inline __bf16 f2bf(float f) { return (__bf16)f; }

// Issue (no wait) a TDM gather: 16 rows (16-bit indices) of `rowlen` bf16 elements
// from a [nrows x rowlen] bf16 tensor at `gbase` into LDS at `lds_off`, padding
// 4 dwords per 256 dwords. Tracked by TENSORcnt (in-order per wave).
__device__ inline void tdm_gather16_issue(unsigned long long gbase, unsigned lds_off,
                                          int rowlen, int nrows, const unsigned* idx16) {
  u32x4 g0; i32x8 g1; i32x4 g2, g3;
  g0[0] = 0x80000001u;                                   // count=1, gather_mode=1, 16b idx
  g0[1] = lds_off;                                       // LDS dest (bytes)
  g0[2] = (unsigned)(gbase & 0xffffffffu);               // global addr lo
  g0[3] = (unsigned)((gbase >> 32) & 0x01ffffffu) | 0x80000000u;  // addr hi | type=2
  g1[0] = 0x07D10000;            // data_size=2B | pad_enable | pad_interval=256dw | pad_amount=4dw
  g1[1] = (int)((unsigned)rowlen << 16);                 // tensor_dim0[15:0]
  g1[2] = (int)((unsigned)nrows << 16);                  // dim0 hi=0 | tensor_dim1[15:0]
  g1[3] = (int)((unsigned)rowlen << 16);                 // dim1 hi=0 | tile_dim0
  g1[4] = 16;                                            // tile_dim1 = #valid indices
  g1[5] = rowlen;                                        // tensor_dim0_stride lo
  g1[6] = 0; g1[7] = 0;
  g2[0] = (int)(idx16[0]  | (idx16[1]  << 16));
  g2[1] = (int)(idx16[2]  | (idx16[3]  << 16));
  g2[2] = (int)(idx16[4]  | (idx16[5]  << 16));
  g2[3] = (int)(idx16[6]  | (idx16[7]  << 16));
  g3[0] = (int)(idx16[8]  | (idx16[9]  << 16));
  g3[1] = (int)(idx16[10] | (idx16[11] << 16));
  g3[2] = (int)(idx16[12] | (idx16[13] << 16));
  g3[3] = (int)(idx16[14] | (idx16[15] << 16));
#if __clang_major__ >= 23
  i32x8 g4 = {0, 0, 0, 0, 0, 0, 0, 0};
  __builtin_amdgcn_tensor_load_to_lds(g0, g1, g2, g3, g4, 0);
#else
  __builtin_amdgcn_tensor_load_to_lds(g0, g1, g2, g3, 0);
#endif
}

__global__ void k_zero_counts(int* cnt) {
  if (threadIdx.x < NEXP) cnt[threadIdx.x] = 0;
}

// One block per token: RMSNorm -> bf16 t, gate logits, top-4 softmax, scatter to expert lists.
__global__ void k_rms_gate_topk(const float* __restrict__ x,
                                const float* __restrict__ norm_scale,
                                const float* __restrict__ gate_w,
                                const float* __restrict__ gate_b,
                                __bf16* __restrict__ tbf,
                                int*   __restrict__ topk_idx,
                                float* __restrict__ topk_w,
                                int*   __restrict__ cnt,
                                int*   __restrict__ list,
                                int T) {
  __shared__ float s_t[HDIM];
  __shared__ float s_red[256];
  __shared__ float s_g[NEXP];
  const int t = blockIdx.x;
  const int tid = threadIdx.x;
  const float* xp = x + (size_t)t * HDIM;

  float ss = 0.f;
  for (int i = tid; i < HDIM; i += 256) { float v = xp[i]; s_t[i] = v; ss += v * v; }
  s_red[tid] = ss;
  __syncthreads();
  for (int off = 128; off > 0; off >>= 1) {
    if (tid < off) s_red[tid] += s_red[tid + off];
    __syncthreads();
  }
  const float rms = rsqrtf(s_red[0] / (float)HDIM + EPSC);

  for (int i = tid; i < HDIM; i += 256) {
    float v = s_t[i] * rms * norm_scale[i];
    s_t[i] = v;
    tbf[(size_t)t * HDIM + i] = f2bf(v);
  }
  __syncthreads();

  if (tid < NEXP) {
    const float* gw = gate_w + (size_t)tid * HDIM;
    float acc = gate_b[tid];
    for (int i = 0; i < HDIM; ++i) acc += s_t[i] * gw[i];
    s_g[tid] = acc;
  }
  __syncthreads();

  if (tid == 0) {
    float vals[KSEL]; int idxs[KSEL];
    bool used[NEXP];
    for (int e = 0; e < NEXP; ++e) used[e] = false;
    for (int k = 0; k < KSEL; ++k) {
      float best = -INFINITY; int bi = 0;
      for (int e = 0; e < NEXP; ++e)
        if (!used[e] && s_g[e] > best) { best = s_g[e]; bi = e; }
      used[bi] = true; vals[k] = best; idxs[k] = bi;
    }
    const float m = vals[0];
    float w[KSEL], sum = 0.f;
    for (int k = 0; k < KSEL; ++k) { w[k] = __expf(vals[k] - m); sum += w[k]; }
    for (int k = 0; k < KSEL; ++k) {
      const float ww = w[k] / sum;
      topk_idx[t * KSEL + k] = idxs[k];
      topk_w[t * KSEL + k]   = ww;
      int pos = atomicAdd(&cnt[idxs[k]], 1);
      list[idxs[k] * T + pos] = t * KSEL + k;   // assignment id
    }
  }
}

// GEMM1 + SwiGLU. grid = NEXP * 8 (8 col-blocks of 256 over 2F). 256 thr = 8 waves,
// each wave a 16x32 tile (2 accs sharing one A frag). A-panel double-buffered via
// TDM gather (issue tile i+1, s_wait_tensorcnt<=1 for tile i, barrier, compute).
__global__ void k_gemm1_swiglu(const __bf16* __restrict__ tbf,
                               const float*  __restrict__ w1,
                               const float*  __restrict__ b1,
                               const int*    __restrict__ cnt,
                               const int*    __restrict__ list,
                               const float*  __restrict__ topk_w,
                               __bf16*       __restrict__ hbuf,
                               int T) {
  __shared__ __bf16 s_a[2][16 * AROW_ELEM];   // 2 x 33280B double-buffered A panel
  __shared__ float s_h1[16][264];             // 16 x 256 (+pad) epilogue tile
  const int e    = blockIdx.x >> 3;
  const int nb   = blockIdx.x & 7;
  const int wave = threadIdx.x >> 5;
  const int lane = threadIdx.x & 31;
  const int half = lane >> 4;
  const int l16  = lane & 15;
  const int Ne   = cnt[e];
  const int n0   = nb * 256 + wave * 32;
  const float* wbase0 = w1 + ((size_t)e * 2 * FDIM + n0 + l16) * HDIM + half * 16;
  const float* wbase1 = wbase0 + (size_t)16 * HDIM;
  const int nt = (Ne + 15) >> 4;

  if (threadIdx.x < 32 && nt > 0) {          // prologue: gather row tile 0 into buf 0
    unsigned idx16[16];
#pragma unroll
    for (int j = 0; j < 16; ++j) {
      int rj = j; if (rj >= Ne) rj = Ne - 1;
      idx16[j] = (unsigned)(list[e * T + rj] >> 2);
    }
    tdm_gather16_issue((unsigned long long)(size_t)tbf,
                       (unsigned)(size_t)&s_a[0][0], HDIM, T, idx16);
  }

  for (int it = 0; it < nt; ++it) {
    const int r0 = it << 4;
    if (threadIdx.x < 32) {
      if (it + 1 < nt) {                     // prefetch next tile into other buffer
        unsigned idx16[16];
#pragma unroll
        for (int j = 0; j < 16; ++j) {
          int rj = r0 + 16 + j; if (rj >= Ne) rj = Ne - 1;
          idx16[j] = (unsigned)(list[e * T + rj] >> 2);
        }
        tdm_gather16_issue((unsigned long long)(size_t)tbf,
                           (unsigned)(size_t)&s_a[(it + 1) & 1][0], HDIM, T, idx16);
        __builtin_amdgcn_s_wait_tensorcnt(1);   // current buffer landed (in-order)
      } else {
        __builtin_amdgcn_s_wait_tensorcnt(0);
      }
    }
    __syncthreads();                         // A panel visible to all 8 waves
    const char* ap = (const char*)&s_a[it & 1][0] + (size_t)l16 * AROW_BYTES;

    v8f acc0, acc1;
#pragma unroll
    for (int i = 0; i < 8; ++i) { acc0[i] = 0.f; acc1[i] = 0.f; }

#pragma unroll 4
    for (int k0 = 0; k0 < HDIM; k0 += 32) {
      const int pb = (k0 >= 512) ? 16 : 0;   // TDM mid-row pad
      v16bf af, bfm0, bfm1;
      bf16x8 lo = *(const bf16x8*)(ap + 2 * (k0 + half * 16) + pb);
      bf16x8 hi = *(const bf16x8*)(ap + 2 * (k0 + 8 + half * 16) + pb);
#pragma unroll
      for (int i = 0; i < 8; ++i) { af[i] = lo[i]; af[8 + i] = hi[i]; }
      const float* wp0 = wbase0 + k0;
      const float* wp1 = wbase1 + k0;
      __builtin_prefetch(wp0 + 256, 0, 1);
      __builtin_prefetch(wp1 + 256, 0, 1);
#pragma unroll
      for (int i = 0; i < 16; ++i) bfm0[i] = f2bf(wp0[i]);
#pragma unroll
      for (int i = 0; i < 16; ++i) bfm1[i] = f2bf(wp1[i]);
      acc0 = __builtin_amdgcn_wmma_f32_16x16x32_bf16(false, af, false, bfm0,
                                                     (short)0, acc0, false, false);
      acc1 = __builtin_amdgcn_wmma_f32_16x16x32_bf16(false, af, false, bfm1,
                                                     (short)0, acc1, false, false);
    }
#pragma unroll
    for (int r = 0; r < 8; ++r) {
      s_h1[r + half * 8][wave * 32 + l16]      = acc0[r];
      s_h1[r + half * 8][wave * 32 + 16 + l16] = acc1[r];
    }
    __syncthreads();

    const int fb = nb * 128;
    for (int q = threadIdx.x; q < 16 * 128; q += 256) {
      const int rr = q >> 7;
      const int fl = q & 127;
      if (r0 + rr < Ne) {
        const int aid2 = list[e * T + r0 + rr];
        const int cg = 2 * fl, cl = 2 * fl + 1;
        float glu = s_h1[rr][cg] + b1[(size_t)e * 2 * FDIM + nb * 256 + cg];
        float lin = s_h1[rr][cl] + b1[(size_t)e * 2 * FDIM + nb * 256 + cl];
        glu = fminf(glu, LIMITC);
        lin = fminf(fmaxf(lin, -LIMITC), LIMITC);
        const float sig = __builtin_amdgcn_rcpf(1.f + __expf(-ALPHAC * glu));
        const float val = glu * sig * (lin + 1.f) * topk_w[aid2];
        hbuf[(size_t)aid2 * FDIM + fb + fl] = f2bf(val);
      }
    }
    __syncthreads();             // protects s_h1 and read side of s_a rotation
  }
}

// GEMM2: h[Ne,F] (bf16, rows = assignment ids, TDM-gathered, double-buffered)
// x w2[e]^T -> fp32 partials per assignment (deterministic combine).
__global__ void k_gemm2(const __bf16* __restrict__ hbuf,
                        const float*  __restrict__ w2,
                        const int*    __restrict__ cnt,
                        const int*    __restrict__ list,
                        float*        __restrict__ partial,
                        int T) {
  __shared__ __bf16 s_a[2][16 * AROW_ELEM];
  const int e    = blockIdx.x >> 2;     // 4 col-blocks of 256 over H
  const int nb   = blockIdx.x & 3;
  const int wave = threadIdx.x >> 5;
  const int lane = threadIdx.x & 31;
  const int half = lane >> 4;
  const int l16  = lane & 15;
  const int Ne   = cnt[e];
  const int n0   = nb * 256 + wave * 32;
  const float* wbase0 = w2 + ((size_t)e * HDIM + n0 + l16) * FDIM + half * 16;
  const float* wbase1 = wbase0 + (size_t)16 * FDIM;
  const int nt = (Ne + 15) >> 4;

  if (threadIdx.x < 32 && nt > 0) {
    unsigned idx16[16];
#pragma unroll
    for (int j = 0; j < 16; ++j) {
      int rj = j; if (rj >= Ne) rj = Ne - 1;
      idx16[j] = (unsigned)list[e * T + rj];
    }
    tdm_gather16_issue((unsigned long long)(size_t)hbuf,
                       (unsigned)(size_t)&s_a[0][0], FDIM, T * KSEL, idx16);
  }

  for (int it = 0; it < nt; ++it) {
    const int r0 = it << 4;
    if (threadIdx.x < 32) {
      if (it + 1 < nt) {
        unsigned idx16[16];
#pragma unroll
        for (int j = 0; j < 16; ++j) {
          int rj = r0 + 16 + j; if (rj >= Ne) rj = Ne - 1;
          idx16[j] = (unsigned)list[e * T + rj];
        }
        tdm_gather16_issue((unsigned long long)(size_t)hbuf,
                           (unsigned)(size_t)&s_a[(it + 1) & 1][0], FDIM, T * KSEL, idx16);
        __builtin_amdgcn_s_wait_tensorcnt(1);
      } else {
        __builtin_amdgcn_s_wait_tensorcnt(0);
      }
    }
    __syncthreads();
    const char* ap = (const char*)&s_a[it & 1][0] + (size_t)l16 * AROW_BYTES;

    v8f acc0, acc1;
#pragma unroll
    for (int i = 0; i < 8; ++i) { acc0[i] = 0.f; acc1[i] = 0.f; }

#pragma unroll 4
    for (int k0 = 0; k0 < FDIM; k0 += 32) {
      const int pb = (k0 >= 512) ? 16 : 0;
      v16bf af, bfm0, bfm1;
      bf16x8 lo = *(const bf16x8*)(ap + 2 * (k0 + half * 16) + pb);
      bf16x8 hi = *(const bf16x8*)(ap + 2 * (k0 + 8 + half * 16) + pb);
#pragma unroll
      for (int i = 0; i < 8; ++i) { af[i] = lo[i]; af[8 + i] = hi[i]; }
      const float* wp0 = wbase0 + k0;
      const float* wp1 = wbase1 + k0;
      __builtin_prefetch(wp0 + 256, 0, 1);
      __builtin_prefetch(wp1 + 256, 0, 1);
#pragma unroll
      for (int i = 0; i < 16; ++i) bfm0[i] = f2bf(wp0[i]);
#pragma unroll
      for (int i = 0; i < 16; ++i) bfm1[i] = f2bf(wp1[i]);
      acc0 = __builtin_amdgcn_wmma_f32_16x16x32_bf16(false, af, false, bfm0,
                                                     (short)0, acc0, false, false);
      acc1 = __builtin_amdgcn_wmma_f32_16x16x32_bf16(false, af, false, bfm1,
                                                     (short)0, acc1, false, false);
    }
#pragma unroll
    for (int r = 0; r < 8; ++r) {
      const int row = r + half * 8;
      if (r0 + row < Ne) {
        const int aid2 = list[e * T + r0 + row];
        partial[(size_t)aid2 * HDIM + n0 + l16]      = acc0[r];
        partial[(size_t)aid2 * HDIM + n0 + 16 + l16] = acc1[r];
      }
    }
    __syncthreads();             // read side of s_a rotation
  }
}

// out = x + sum_k partial[t,k] + sum_k w_k * b2[e_k]
__global__ void k_finalize(const float* __restrict__ x,
                           const float* __restrict__ partial,
                           const int*   __restrict__ topk_idx,
                           const float* __restrict__ topk_w,
                           const float* __restrict__ b2,
                           float*       __restrict__ out,
                           int T) {
  const int t = blockIdx.x;
  const int ti4 = t * KSEL;
  int ee[KSEL]; float ww[KSEL];
#pragma unroll
  for (int k = 0; k < KSEL; ++k) { ee[k] = topk_idx[ti4 + k]; ww[k] = topk_w[ti4 + k]; }
#pragma unroll
  for (int j = 0; j < 4; ++j) {
    const int h = threadIdx.x * 4 + j;
    float v = x[(size_t)t * HDIM + h];
#pragma unroll
    for (int k = 0; k < KSEL; ++k) {
      v += partial[(size_t)(ti4 + k) * HDIM + h];
      v += ww[k] * b2[(size_t)ee[k] * HDIM + h];
    }
    out[(size_t)t * HDIM + h] = v;
  }
}

extern "C" void kernel_launch(void* const* d_in, const int* in_sizes, int n_in,
                              void* d_out, int out_size, void* d_ws, size_t ws_size,
                              hipStream_t stream) {
  const float* x          = (const float*)d_in[0];
  const float* norm_scale = (const float*)d_in[1];
  const float* gate_w     = (const float*)d_in[2];
  const float* gate_b     = (const float*)d_in[3];
  const float* w1         = (const float*)d_in[4];
  const float* b1         = (const float*)d_in[5];
  const float* w2         = (const float*)d_in[6];
  const float* b2         = (const float*)d_in[7];
  float* out = (float*)d_out;

  const int T = in_sizes[0] / HDIM;   // tokens (B*S)

  char* ws = (char*)d_ws;
  size_t off = 0;
  __bf16* tbf     = (__bf16*)(ws + off); off += (size_t)T * HDIM * sizeof(__bf16);
  int*    tk_idx  = (int*)  (ws + off); off += (size_t)T * KSEL * sizeof(int);
  float*  tk_w    = (float*)(ws + off); off += (size_t)T * KSEL * sizeof(float);
  int*    cnt     = (int*)  (ws + off); off += 128;
  int*    list    = (int*)  (ws + off); off += (size_t)NEXP * T * sizeof(int);
  __bf16* hbuf    = (__bf16*)(ws + off); off += (size_t)T * KSEL * FDIM * sizeof(__bf16);
  float*  partial = (float*)(ws + off); off += (size_t)T * KSEL * HDIM * sizeof(float);
  (void)ws_size; (void)n_in; (void)out_size;

  k_zero_counts<<<1, 32, 0, stream>>>(cnt);
  k_rms_gate_topk<<<T, 256, 0, stream>>>(x, norm_scale, gate_w, gate_b,
                                         tbf, tk_idx, tk_w, cnt, list, T);
  k_gemm1_swiglu<<<NEXP * 8, 256, 0, stream>>>(tbf, w1, b1, cnt, list, tk_w, hbuf, T);
  k_gemm2<<<NEXP * 4, 256, 0, stream>>>(hbuf, w2, cnt, list, partial, T);
  k_finalize<<<T, 256, 0, stream>>>(x, partial, tk_idx, tk_w, b2, out, T);
}
